// RGNN_34600256537089
// MI455X (gfx1250) — matrix operand order repaired and checked by reading, not compiled, not run on hardware
//
#include <hip/hip_runtime.h>
#include <hip/hip_bf16.h>
#include <cstdint>

// ---------------------------------------------------------------------------
// RGNN (2-layer hetero GAT) for MI455X / gfx1250.
//   - Dense projections: bf16 WMMA (v_wmma_f32_16x16x32_bf16), f32 accum.
//   - A tiles: global_load_async_to_lds_b128 (ASYNCcnt, no VGPR staging).
//   - B tiles: TDM tensor_load_to_lds with hardware LDS padding (TENSORcnt).
//   - Edge softmax: float atomic max/add passes; message scatter: f32 atomics.
//   - Layer-1 'pa' conv is dead code in the reference (out_a unused) -> skipped.
// ---------------------------------------------------------------------------

typedef __bf16 bf16_t;
typedef __attribute__((ext_vector_type(16))) __bf16 v16bf;
typedef __attribute__((ext_vector_type(8)))  __bf16 v8bf;
typedef __attribute__((ext_vector_type(8)))  float  v8f;
typedef __attribute__((ext_vector_type(4)))  int    v4i;
typedef __attribute__((ext_vector_type(8)))  int    v8i;

#define N_PAPER  50000
#define N_AUTHOR 50000
#define NE       300000
#define IN_DIM   1024
#define H_DIM    256
#define OUT_DIM  19

// ------------------------------- GEMM --------------------------------------
// C[M x 256] = A[M x K](bf16, row-major) @ Wt[256 x K](bf16, N-major rows)
// Block: 256 threads = 8 waves (4x2), wave does 2x2 wmma tiles -> 128x64/block.
#define BM 128
#define BN 64
#define BK 32
#define ASTRIDE 40   // elements; 80B row stride -> conflict-free b128 lane reads
#define BSTRIDE 40   // matches TDM pad: 64B data + 16B pad per row

union FragBF { v16bf v; v8bf h[2]; };

__global__ __launch_bounds__(256)
void gemm_bf16_wmma(const bf16_t* __restrict__ A, const bf16_t* __restrict__ Bt,
                    float* __restrict__ C, int M, int K) {
  __shared__ bf16_t As[BM * ASTRIDE];
  __shared__ bf16_t Bs[BN * BSTRIDE];

  const int tid    = threadIdx.x;
  const int lane   = tid & 31;
  const int wave   = tid >> 5;
  const int laneLo = lane & 15;
  const int laneHi = lane >> 4;          // 0 or 1
  const int wm     = wave & 3;           // 0..3  -> 32-row strip
  const int wn     = wave >> 2;          // 0..1  -> 32-col strip
  const int blockM = blockIdx.x * BM;
  const int blockN = blockIdx.y * BN;

  // ---- A-tile async-copy assignment: 512 x 16B chunks, 2 per thread ----
  // Chunk ch -> LDS row r = ch/4, elem off = (ch%4)*8. LDS dest is constant
  // across K-steps; only the global address advances.
  const int ch0 = tid, ch1 = tid + 256;
  const int r0 = ch0 >> 2, o0 = (ch0 & 3) * 8;
  const int r1 = ch1 >> 2, o1 = (ch1 & 3) * 8;
  int gr0 = blockM + r0; if (gr0 >= M) gr0 = M - 1;
  int gr1 = blockM + r1; if (gr1 >= M) gr1 = M - 1;
  const bf16_t* gA0 = A + (size_t)gr0 * K + o0;
  const bf16_t* gA1 = A + (size_t)gr1 * K + o1;
  const unsigned lA0 = (unsigned)(uintptr_t)(As + r0 * ASTRIDE + o0);
  const unsigned lA1 = (unsigned)(uintptr_t)(As + r1 * ASTRIDE + o1);
  const unsigned lB  = (unsigned)(uintptr_t)(Bs);

  v8f acc[2][2] = {};

  const int KT = K / BK;
  for (int kt = 0; kt < KT; ++kt) {
    const int kk = kt * BK;
    __syncthreads();   // all waves done reading previous LDS tiles

    // ---- B tile via Tensor Data Mover (wave 0 issues block-wide DMA) ----
    // D# 2-D tile: 64 rows (N) x 32 elems (K), element 2B, row stride K elems.
    // pad_enable: pad 4 DWORDs every 16 DWORDs -> 80B LDS row stride.
    if (wave == 0) {
      unsigned long long ga =
          (unsigned long long)(uintptr_t)(Bt + (size_t)blockN * K + kk);
      v4i g0;
      g0[0] = 1;                                    // count=1, user mode
      g0[1] = (int)lB;                              // lds_addr
      g0[2] = (int)(unsigned)(ga & 0xffffffffu);    // global_addr[31:0]
      g0[3] = (int)((((unsigned)(ga >> 32)) & 0x01ffffffu) | 0x80000000u); // type=2
      v8i g1;
      g1[0] = (1 << 16)      // data_size = 2B
            | (1 << 20)      // pad_enable
            | (3 << 22)      // pad_interval: 16 DWORDs (64B)
            | (3 << 25);     // pad_amount:   4 DWORDs (16B)
      g1[1] = (int)((unsigned)K << 16);                         // tensor_dim0 lo16
      g1[2] = (int)((((unsigned)K >> 16) & 0xffffu) | (64u << 16)); // td0 hi | tensor_dim1=64
      g1[3] = (int)(32u << 16);                                 // tile_dim0 = 32
      g1[4] = 64;                                               // tile_dim1 = 64
      g1[5] = K;                                                // tensor_dim0_stride lo32
      g1[6] = 0;
      g1[7] = 0;
      asm volatile("tensor_load_to_lds %0, %1" :: "s"(g0), "s"(g1) : "memory");
    }

    // ---- A tile via per-lane async global->LDS b128 ----
    asm volatile("global_load_async_to_lds_b128 %0, %1, off"
                 :: "v"(lA0), "v"(gA0 + kk) : "memory");
    asm volatile("global_load_async_to_lds_b128 %0, %1, off"
                 :: "v"(lA1), "v"(gA1 + kk) : "memory");

    asm volatile("s_wait_asynccnt 0x0" ::: "memory");
    if (wave == 0) __builtin_amdgcn_s_wait_tensorcnt((short)0);
    __syncthreads();   // publish LDS tiles to all waves

    // ---- build fragments (CDNA5 wave32 WMMA VGPR layouts) ----
    FragBF a[2], b[2];
#pragma unroll
    for (int i = 0; i < 2; ++i) {
      int row = wm * 32 + i * 16 + laneLo;
      // lanes 0-15: elems0-7 = K0-7, elems8-15 = K16-23; lanes 16-31: +8
      a[i].h[0] = *(const v8bf*)(As + row * ASTRIDE + laneHi * 8);
      a[i].h[1] = *(const v8bf*)(As + row * ASTRIDE + 16 + laneHi * 8);
    }
#pragma unroll
    for (int j = 0; j < 2; ++j) {
      int col = wn * 32 + j * 16 + laneLo;
      // lanes 0-15: elems0-7 = K0-7, elems8-15 = K16-23; lanes 16-31: +8
      b[j].h[0] = *(const v8bf*)(Bs + col * BSTRIDE + laneHi * 8);
      b[j].h[1] = *(const v8bf*)(Bs + col * BSTRIDE + 16 + laneHi * 8);
    }
#pragma unroll
    for (int i = 0; i < 2; ++i)
#pragma unroll
      for (int j = 0; j < 2; ++j)
        acc[i][j] = __builtin_amdgcn_wmma_f32_16x16x32_bf16(
            false, a[i].v, false, b[j].v, (short)0, acc[i][j], false, false);
  }

  // ---- store: C/D layout: elem r -> row = r + 8*laneHi, col = laneLo ----
#pragma unroll
  for (int i = 0; i < 2; ++i) {
    int rowBase = blockM + wm * 32 + i * 16 + laneHi * 8;
#pragma unroll
    for (int j = 0; j < 2; ++j) {
      int col = blockN + wn * 32 + j * 16 + laneLo;
#pragma unroll
      for (int r = 0; r < 8; ++r) {
        int row = rowBase + r;
        if (row < M) C[(size_t)row * 256 + col] = acc[i][j][r];
      }
    }
  }
}

// --------------------------- small utility kernels -------------------------
__global__ void f32_to_bf16_k(const float* __restrict__ in, bf16_t* __restrict__ out, long n) {
  long t = (long)blockIdx.x * blockDim.x + threadIdx.x;
  if (t < n) out[t] = (bf16_t)in[t];
}

// W [K x 256] f32 -> Wt [256 x K] bf16 (transposed so GEMM B loads contiguously)
__global__ void w_transpose_bf16_k(const float* __restrict__ W, bf16_t* __restrict__ Wt, int K) {
  long t = (long)blockIdx.x * blockDim.x + threadIdx.x;
  long n = (long)K * 256;
  if (t >= n) return;
  int k = (int)(t >> 8), c = (int)(t & 255);
  Wt[(size_t)c * K + k] = (bf16_t)W[t];
}

__global__ void fill_f32_k(float* __restrict__ p, float v, long n) {
  long t = (long)blockIdx.x * blockDim.x + threadIdx.x;
  if (t < n) p[t] = v;
}

// out[n,d] = b1[d] + (b2 ? b2[d] : 0)   (bias sum across edge types on dst)
__global__ void init_bias_k(float* __restrict__ out, const float* __restrict__ b1,
                            const float* __restrict__ b2, long n) {
  long t = (long)blockIdx.x * blockDim.x + threadIdx.x;
  if (t >= n) return;
  int d = (int)(t & 255);
  out[t] = b1[d] + (b2 ? b2[d] : 0.f);
}

// a[n,h] = sum_c H[n, h*64+c] * att[h*64+c]
__global__ void att_score_k(const float* __restrict__ H, const float* __restrict__ att,
                            float* __restrict__ out, int N) {
  int t = blockIdx.x * blockDim.x + threadIdx.x;
  if (t >= N * 4) return;
  int n = t >> 2, h = t & 3;
  const float* hp = H + (size_t)n * 256 + h * 64;
  const float* ap = att + h * 64;
  float s = 0.f;
#pragma unroll 8
  for (int c = 0; c < 64; ++c) s += hp[c] * ap[c];
  out[t] = s;
}

__device__ __forceinline__ float lrelu02(float x) { return x > 0.f ? x : 0.2f * x; }

__device__ __forceinline__ void atomicMaxF(float* addr, float val) {
  if (val >= 0.f) atomicMax((int*)addr, __float_as_int(val));
  else            atomicMin((unsigned int*)addr, __float_as_uint(val));
}

__global__ void edge_max_k(const int* __restrict__ src, const int* __restrict__ dst,
                           const float* __restrict__ as, const float* __restrict__ ad,
                           float* __restrict__ m, int E) {
  int t = blockIdx.x * blockDim.x + threadIdx.x;
  if (t >= E * 4) return;
  int e = t >> 2, h = t & 3;
  int s = src[e], d = dst[e];
  float x = lrelu02(as[s * 4 + h] + ad[d * 4 + h]);
  atomicMaxF(&m[d * 4 + h], x);
}

__global__ void edge_exp_k(const int* __restrict__ src, const int* __restrict__ dst,
                           const float* __restrict__ as, const float* __restrict__ ad,
                           const float* __restrict__ m, float* __restrict__ ex,
                           float* __restrict__ den, int E) {
  int t = blockIdx.x * blockDim.x + threadIdx.x;
  if (t >= E * 4) return;
  int e = t >> 2, h = t & 3;
  int s = src[e], d = dst[e];
  float x = lrelu02(as[s * 4 + h] + ad[d * 4 + h]);
  float v = __expf(x - m[d * 4 + h]);
  ex[t] = v;
  atomicAdd(&den[d * 4 + h], v);
}

// 64 threads per edge, 4 channels each: out[dst] += alpha * Hs[src]
__global__ void edge_scatter_k(const int* __restrict__ src, const int* __restrict__ dst,
                               const float* __restrict__ ex, const float* __restrict__ den,
                               const float* __restrict__ Hs, float* __restrict__ out, int E) {
  long t = (long)blockIdx.x * blockDim.x + threadIdx.x;
  if (t >= (long)E * 64) return;
  int e  = (int)(t >> 6);
  int c0 = (int)(t & 63) * 4;     // 0..252
  int h  = c0 >> 6;
  int s = src[e], d = dst[e];
  float alpha = ex[e * 4 + h] / (den[d * 4 + h] + 1e-16f);
  float4 hv = *(const float4*)(Hs + (size_t)s * 256 + c0);
  float* o = out + (size_t)d * 256 + c0;
  atomicAdd(o + 0, alpha * hv.x);
  atomicAdd(o + 1, alpha * hv.y);
  atomicAdd(o + 2, alpha * hv.z);
  atomicAdd(o + 3, alpha * hv.w);
}

// leaky_relu(0.01) then down-convert for next layer's bf16 GEMM input
__global__ void act_to_bf16_k(const float* __restrict__ in, bf16_t* __restrict__ out, long n) {
  long t = (long)blockIdx.x * blockDim.x + threadIdx.x;
  if (t >= n) return;
  float x = in[t];
  out[t] = (bf16_t)(x > 0.f ? x : 0.01f * x);
}

// out[n,:19] = X[n,:256] @ Wl[256,19] + bl
__global__ __launch_bounds__(64)
void final_linear_k(const float* __restrict__ X, const float* __restrict__ Wl,
                    const float* __restrict__ bl, float* __restrict__ out, int N) {
  __shared__ float row[256];
  int n = blockIdx.x;
  for (int i = threadIdx.x; i < 256; i += 64) row[i] = X[(size_t)n * 256 + i];
  __syncthreads();
  if (threadIdx.x < OUT_DIM) {
    float s = bl[threadIdx.x];
#pragma unroll 8
    for (int k = 0; k < 256; ++k) s += row[k] * Wl[k * OUT_DIM + threadIdx.x];
    out[(size_t)n * OUT_DIM + threadIdx.x] = s;
  }
}

// ------------------------------ host side ----------------------------------
static inline int cdiv(long a, long b) { return (int)((a + b - 1) / b); }

// One GATConv: projections -> Hs/Hd; then attention + softmax + scatter.
static void run_conv(hipStream_t st,
                     const bf16_t* Xs, int Ns, const bf16_t* Xd, int Nd, int K,
                     const bf16_t* Wt, const float* att_src, const float* att_dst,
                     const int* edges, int E,
                     float* Hs, float* Hd,
                     float* asrc, float* adst, float* mbuf, float* den, float* ex,
                     float* out) {
  dim3 gs(cdiv(Ns, BM), 256 / BN);
  gemm_bf16_wmma<<<gs, 256, 0, st>>>(Xs, Wt, Hs, Ns, K);
  if (Hd != Hs) {
    dim3 gd(cdiv(Nd, BM), 256 / BN);
    gemm_bf16_wmma<<<gd, 256, 0, st>>>(Xd, Wt, Hd, Nd, K);
  }
  att_score_k<<<cdiv((long)Ns * 4, 256), 256, 0, st>>>(Hs, att_src, asrc, Ns);
  att_score_k<<<cdiv((long)Nd * 4, 256), 256, 0, st>>>(Hd, att_dst, adst, Nd);
  fill_f32_k<<<cdiv((long)Nd * 4, 256), 256, 0, st>>>(mbuf, -__builtin_inff(), (long)Nd * 4);
  fill_f32_k<<<cdiv((long)Nd * 4, 256), 256, 0, st>>>(den, 0.f, (long)Nd * 4);
  const int* src = edges;
  const int* dst = edges + E;
  edge_max_k<<<cdiv((long)E * 4, 256), 256, 0, st>>>(src, dst, asrc, adst, mbuf, E);
  edge_exp_k<<<cdiv((long)E * 4, 256), 256, 0, st>>>(src, dst, asrc, adst, mbuf, ex, den, E);
  edge_scatter_k<<<cdiv((long)E * 64, 256), 256, 0, st>>>(src, dst, ex, den, Hs, out, E);
}

extern "C" void kernel_launch(void* const* d_in, const int* in_sizes, int n_in,
                              void* d_out, int out_size, void* d_ws, size_t ws_size,
                              hipStream_t stream) {
  (void)in_sizes; (void)n_in; (void)out_size; (void)ws_size;
  // JAX pytree (sorted-key) flatten order:
  //  0 x_paper  1 x_author  2 ei_pp  3 ei_ap  4 ei_pa
  //  layers[0]: ap(W,att_dst,att_src,bias)=5..8  pa=9..12  pp=13..16
  //  layers[1]: ap=17..20  pa=21..24  pp=25..28
  //  29 lin_b   30 lin_w
  const float* xp     = (const float*)d_in[0];
  const float* xa     = (const float*)d_in[1];
  const int*   ei_pp  = (const int*)d_in[2];
  const int*   ei_ap  = (const int*)d_in[3];
  const int*   ei_pa  = (const int*)d_in[4];
  const float* L0ap_W  = (const float*)d_in[5];
  const float* L0ap_ad = (const float*)d_in[6];
  const float* L0ap_as = (const float*)d_in[7];
  const float* L0ap_b  = (const float*)d_in[8];
  const float* L0pa_W  = (const float*)d_in[9];
  const float* L0pa_ad = (const float*)d_in[10];
  const float* L0pa_as = (const float*)d_in[11];
  const float* L0pa_b  = (const float*)d_in[12];
  const float* L0pp_W  = (const float*)d_in[13];
  const float* L0pp_ad = (const float*)d_in[14];
  const float* L0pp_as = (const float*)d_in[15];
  const float* L0pp_b  = (const float*)d_in[16];
  const float* L1ap_W  = (const float*)d_in[17];
  const float* L1ap_ad = (const float*)d_in[18];
  const float* L1ap_as = (const float*)d_in[19];
  const float* L1ap_b  = (const float*)d_in[20];
  const float* L1pp_W  = (const float*)d_in[25];
  const float* L1pp_ad = (const float*)d_in[26];
  const float* L1pp_as = (const float*)d_in[27];
  const float* L1pp_b  = (const float*)d_in[28];
  const float* lin_b  = (const float*)d_in[29];
  const float* lin_w  = (const float*)d_in[30];
  float* out = (float*)d_out;

  // ---- workspace carve-up (stream-serialized; everything init'd per call) ----
  char* base = (char*)d_ws;
  size_t off = 0;
  auto alloc = [&](size_t bytes) -> char* {
    char* p = base + off;
    off = (off + bytes + 255) & ~(size_t)255;
    return p;
  };
  bf16_t* XPB   = (bf16_t*)alloc((size_t)N_PAPER * IN_DIM * 2);
  bf16_t* XAB   = (bf16_t*)alloc((size_t)N_AUTHOR * IN_DIM * 2);
  bf16_t* WT0pp = (bf16_t*)alloc((size_t)IN_DIM * 256 * 2);
  bf16_t* WT0ap = (bf16_t*)alloc((size_t)IN_DIM * 256 * 2);
  bf16_t* WT0pa = (bf16_t*)alloc((size_t)IN_DIM * 256 * 2);
  bf16_t* WT1pp = (bf16_t*)alloc((size_t)H_DIM * 256 * 2);
  bf16_t* WT1ap = (bf16_t*)alloc((size_t)H_DIM * 256 * 2);
  float*  HS    = (float*)alloc((size_t)N_PAPER * 256 * 4);
  float*  HD    = (float*)alloc((size_t)N_PAPER * 256 * 4);
  float*  OUTP  = (float*)alloc((size_t)N_PAPER * 256 * 4);
  float*  OUTA  = (float*)alloc((size_t)N_AUTHOR * 256 * 4);
  bf16_t* X2PB  = (bf16_t*)alloc((size_t)N_PAPER * 256 * 2);
  bf16_t* X2AB  = (bf16_t*)alloc((size_t)N_AUTHOR * 256 * 2);
  float*  ASRC  = (float*)alloc((size_t)N_PAPER * 4 * 4);
  float*  ADST  = (float*)alloc((size_t)N_PAPER * 4 * 4);
  float*  MAXB  = (float*)alloc((size_t)N_PAPER * 4 * 4);
  float*  DEN   = (float*)alloc((size_t)N_PAPER * 4 * 4);
  float*  EXB   = (float*)alloc((size_t)NE * 4 * 4);

  // ---- precision prep: bf16 activations + transposed bf16 weights ----
  long nx = (long)N_PAPER * IN_DIM;
  f32_to_bf16_k<<<cdiv(nx, 256), 256, 0, stream>>>(xp, XPB, nx);
  f32_to_bf16_k<<<cdiv(nx, 256), 256, 0, stream>>>(xa, XAB, nx);
  long nw0 = (long)IN_DIM * 256, nw1 = (long)H_DIM * 256;
  w_transpose_bf16_k<<<cdiv(nw0, 256), 256, 0, stream>>>(L0pp_W, WT0pp, IN_DIM);
  w_transpose_bf16_k<<<cdiv(nw0, 256), 256, 0, stream>>>(L0ap_W, WT0ap, IN_DIM);
  w_transpose_bf16_k<<<cdiv(nw0, 256), 256, 0, stream>>>(L0pa_W, WT0pa, IN_DIM);
  w_transpose_bf16_k<<<cdiv(nw1, 256), 256, 0, stream>>>(L1pp_W, WT1pp, H_DIM);
  w_transpose_bf16_k<<<cdiv(nw1, 256), 256, 0, stream>>>(L1ap_W, WT1ap, H_DIM);

  long nout = (long)N_PAPER * 256;

  // ================= Layer 0 =================
  init_bias_k<<<cdiv(nout, 256), 256, 0, stream>>>(OUTP, L0pp_b, L0ap_b, nout);
  init_bias_k<<<cdiv(nout, 256), 256, 0, stream>>>(OUTA, L0pa_b, nullptr, nout);

  // pp: src=papers, dst=papers (shared projection)
  run_conv(stream, XPB, N_PAPER, XPB, N_PAPER, IN_DIM, WT0pp, L0pp_as, L0pp_ad,
           ei_pp, NE, HS, HS, ASRC, ADST, MAXB, DEN, EXB, OUTP);
  // ap: src=authors, dst=papers
  run_conv(stream, XAB, N_AUTHOR, XPB, N_PAPER, IN_DIM, WT0ap, L0ap_as, L0ap_ad,
           ei_ap, NE, HS, HD, ASRC, ADST, MAXB, DEN, EXB, OUTP);
  // pa: src=papers, dst=authors
  run_conv(stream, XPB, N_PAPER, XAB, N_AUTHOR, IN_DIM, WT0pa, L0pa_as, L0pa_ad,
           ei_pa, NE, HS, HD, ASRC, ADST, MAXB, DEN, EXB, OUTA);

  act_to_bf16_k<<<cdiv(nout, 256), 256, 0, stream>>>(OUTP, X2PB, nout);
  act_to_bf16_k<<<cdiv(nout, 256), 256, 0, stream>>>(OUTA, X2AB, nout);

  // ================= Layer 1 ================= ('pa' dead: out_a unused)
  init_bias_k<<<cdiv(nout, 256), 256, 0, stream>>>(OUTP, L1pp_b, L1ap_b, nout);
  run_conv(stream, X2PB, N_PAPER, X2PB, N_PAPER, H_DIM, WT1pp, L1pp_as, L1pp_ad,
           ei_pp, NE, HS, HS, ASRC, ADST, MAXB, DEN, EXB, OUTP);
  run_conv(stream, X2AB, N_AUTHOR, X2PB, N_PAPER, H_DIM, WT1ap, L1ap_as, L1ap_ad,
           ei_ap, NE, HS, HD, ASRC, ADST, MAXB, DEN, EXB, OUTP);

  // ================= Final linear =================
  final_linear_k<<<N_PAPER, 64, 0, stream>>>(OUTP, lin_w, lin_b, out, N_PAPER);
}
